// LSTMFS_26456998543431
// MI455X (gfx1250) — matrix operand order repaired and checked by reading, not compiled, not run on hardware
//
#include <hip/hip_runtime.h>

#define USE_ASYNC 1

typedef _Float16 f16;
typedef _Float16 v16h __attribute__((ext_vector_type(16)));
typedef _Float16 v8h  __attribute__((ext_vector_type(8)));
typedef float    v8f  __attribute__((ext_vector_type(8)));

#define B_  128
#define S_  256
#define E_  512
#define H_  1024
#define O_  10
#define BH  (B_*H_)            /* 131072 */
#define N4H (4*H_)             /* 4096   */
#define KSPLIT 4               /* split-K factor for recurrent GEMM */

union FragU { v16h v; v8h h[2]; };

// ---- LDS address for async-DMA ops (generic -> AS(3) -> 32-bit LDS offset) ----
__device__ __forceinline__ unsigned lds_off(const f16* p) {
  return (unsigned)(unsigned long long)(const __attribute__((address_space(3))) f16*)p;
}

__device__ __forceinline__ void wait_asynccnt0() {
#if USE_ASYNC
  asm volatile("s_wait_asynccnt 0x0" ::: "memory");
#endif
}

// ---- WMMA fragment loads from LDS tiles (128 rows x 32 K, row-major f16) ----
__device__ __forceinline__ v16h frag_a(const f16* As, int m0, int lane) {
  int row = m0 + (lane & 15);
  int kb  = (lane < 16) ? 0 : 8;          // ISA 16-bit A 16x32 layout
  FragU u;
  u.h[0] = *(const v8h*)(As + row * 32 + kb);
  u.h[1] = *(const v8h*)(As + row * 32 + kb + 16);
  return u.v;
}
__device__ __forceinline__ v16h frag_b(const f16* Bs, int n0, int lane) {
  int col = n0 + (lane & 15);
  int kb  = (lane < 16) ? 0 : 8;          // B 32x16: lanes 0-15 K=0..7,16..23; 16-31 K=8..15,24..31
  FragU u;
  u.h[0] = *(const v8h*)(Bs + col * 32 + kb);
  u.h[1] = *(const v8h*)(Bs + col * 32 + kb + 16);
  return u.v;
}

// ---- stage one 128x32 A tile + 128x32 B tile into LDS (async DMA or sync) ----
__device__ __forceinline__ void issue_tile(const f16* __restrict__ Ag,
                                           const f16* __restrict__ Bg,
                                           int lda, int m_base, int n_base, int k0,
                                           f16* As, f16* Bs, int tid) {
  #pragma unroll
  for (int p = 0; p < 2; ++p) {           // 256 thr x 2 x 16B per tile
    int c   = tid + p * 256;
    int row = c >> 2;
    int ch  = (c & 3) * 8;
    const f16* ga = Ag + (size_t)(m_base + row) * lda + k0 + ch;
    const f16* gb = Bg + (size_t)(n_base + row) * lda + k0 + ch;
    f16* la = As + row * 32 + ch;
    f16* lb = Bs + row * 32 + ch;
#if USE_ASYNC
    asm volatile("global_load_async_to_lds_b128 %0, %1, off"
                 :: "v"(lds_off(la)), "v"(ga) : "memory");
    asm volatile("global_load_async_to_lds_b128 %0, %1, off"
                 :: "v"(lds_off(lb)), "v"(gb) : "memory");
#else
    *(uint4*)la = *(const uint4*)ga;
    *(uint4*)lb = *(const uint4*)gb;
#endif
  }
}

// ---- double-buffered 128x128 tile GEMM core: C += A[M,K] * B[N,K]^T ----
__device__ __forceinline__ void gemm_core(const f16* __restrict__ Ag,
                                          const f16* __restrict__ Bg,
                                          int lda, int Klen, int m_base, int n_base,
                                          f16 (*As)[128 * 32], f16 (*Bs)[128 * 32],
                                          v8f acc[2][4]) {
  const int tid  = threadIdx.x;
  const int lane = tid & 31;
  const int wave = tid >> 5;
  const int wm   = (wave & 3) * 32;   // 4 wave rows  -> 32 M each
  const int wn   = (wave >> 2) * 64;  // 2 wave cols  -> 64 N each
  const int nk   = Klen / 32;

  issue_tile(Ag, Bg, lda, m_base, n_base, 0, As[0], Bs[0], tid);

  for (int kt = 0; kt < nk; ++kt) {
    wait_asynccnt0();                 // tile kt DMA complete (this thread's ops)
    __syncthreads();                  // ... and everyone else's; LDS visible
    if (kt + 1 < nk)                  // overlap next tile's DMA with this tile's math
      issue_tile(Ag, Bg, lda, m_base, n_base, (kt + 1) * 32,
                 As[(kt + 1) & 1], Bs[(kt + 1) & 1], tid);
    const f16* Ac = As[kt & 1];
    const f16* Bc = Bs[kt & 1];
    v16h a[2], b[4];
    #pragma unroll
    for (int i = 0; i < 2; ++i) a[i] = frag_a(Ac, wm + i * 16, lane);
    #pragma unroll
    for (int j = 0; j < 4; ++j) b[j] = frag_b(Bc, wn + j * 16, lane);
    #pragma unroll
    for (int i = 0; i < 2; ++i)
      #pragma unroll
      for (int j = 0; j < 4; ++j)
        acc[i][j] = __builtin_amdgcn_wmma_f32_16x16x32_f16(
            false, a[i], false, b[j], (short)0, acc[i][j], false, false);
    __syncthreads();                  // all waves done reading buf[kt&1]
  }
}

// ---- pack weights (f32 -> f16) + biases into gate-major blocks ----
__global__ void pack_kernel(const float* Wii, const float* Wif, const float* Wig, const float* Wio,
                            const float* Whi, const float* Whf, const float* Whg, const float* Who,
                            const float* bii, const float* bif, const float* big, const float* bio,
                            f16* Wx16, f16* Wh16, float* bx) {
  int idx = blockIdx.x * 256 + threadIdx.x;
  const int WXN = 4 * H_ * E_;          // 2,097,152
  const int WHN = 4 * H_ * H_;          // 4,194,304
  if (idx < WXN) {
    int g = idx / (H_ * E_), off = idx % (H_ * E_);
    const float* s = (g == 0) ? Wii : (g == 1) ? Wif : (g == 2) ? Wig : Wio;
    Wx16[idx] = (f16)s[off];
  } else if (idx < WXN + WHN) {
    int r = idx - WXN;
    int g = r / (H_ * H_), off = r % (H_ * H_);
    const float* s = (g == 0) ? Whi : (g == 1) ? Whf : (g == 2) ? Whg : Who;
    Wh16[r] = (f16)s[off];
  } else if (idx < WXN + WHN + N4H) {
    int r = idx - WXN - WHN;
    int g = r >> 10, off = r & (H_ - 1);
    const float* s = (g == 0) ? bii : (g == 1) ? bif : (g == 2) ? big : bio;
    bx[r] = s[off];
  }
}

// ---- embedding gather + f32->f16 ----
__global__ void embed_kernel(const int* __restrict__ x, const float* __restrict__ emb,
                             f16* __restrict__ e16) {
  size_t idx = (size_t)blockIdx.x * 256 + threadIdx.x;   // B*S*E = 16,777,216
  int r = (int)(idx >> 9);           // row = b*S+s
  int e = (int)(idx & (E_ - 1));
  int tok = x[r];
  e16[idx] = (f16)emb[(size_t)tok * E_ + e];
}

__global__ void init_state_kernel(float* c, float* h32, f16* h16) {
  int i = blockIdx.x * 256 + threadIdx.x;                // BH
  c[i] = 0.f; h32[i] = 0.f; h16[i] = (f16)0.f;
}

// ---- phase 1: all 4 input projections: gx[s][g][b][h] = e @ Wx^T + bx (f16 out) ----
__global__ void gemm_x_kernel(const f16* __restrict__ e16, const f16* __restrict__ Wx16,
                              const float* __restrict__ bx, f16* __restrict__ gx) {
  __shared__ f16 As[2][128 * 32];
  __shared__ f16 Bs[2][128 * 32];
  v8f acc[2][4] = {};
  const int m_base = blockIdx.y * 128;       // 256 tiles over B*S
  const int n_base = blockIdx.x * 128;       // 32 tiles over 4H
  gemm_core(e16, Wx16, E_, E_, m_base, n_base, As, Bs, acc);
  const int lane = threadIdx.x & 31, wave = threadIdx.x >> 5;
  const int wm = (wave & 3) * 32, wn = (wave >> 2) * 64;
  #pragma unroll
  for (int j = 0; j < 4; ++j) {
    int col = n_base + wn + j * 16 + (lane & 15);
    float bias = bx[col];
    int g = col >> 10, h = col & (H_ - 1);
    #pragma unroll
    for (int i = 0; i < 2; ++i)
      #pragma unroll
      for (int r = 0; r < 8; ++r) {
        int row = m_base + wm + i * 16 + r + ((lane < 16) ? 0 : 8);
        int b = row >> 8, s = row & (S_ - 1);
        gx[(((size_t)s * 4 + g) * B_ + b) * H_ + h] = (f16)(acc[i][j][r] + bias);
      }
  }
}

// ---- per-step recurrent GEMM, split-K: pp[kz][g][b][h] = h16 @ Wh^T (partial) ----
__global__ void gemm_h_kernel(const f16* __restrict__ h16, const f16* __restrict__ Wh16,
                              float* __restrict__ pre_part) {
  __shared__ f16 As[2][128 * 32];
  __shared__ f16 Bs[2][128 * 32];
  v8f acc[2][4] = {};
  const int n_base = blockIdx.x * 128;                 // 32 tiles over 4H; M=128
  const int kz     = blockIdx.y;                       // 0..KSPLIT-1
  const int kofs   = kz * (H_ / KSPLIT);               // 256-wide K slice
  gemm_core(h16 + kofs, Wh16 + kofs, H_, H_ / KSPLIT, 0, n_base, As, Bs, acc);
  float* dst = pre_part + (size_t)kz * 4 * BH;
  const int lane = threadIdx.x & 31, wave = threadIdx.x >> 5;
  const int wm = (wave & 3) * 32, wn = (wave >> 2) * 64;
  #pragma unroll
  for (int j = 0; j < 4; ++j) {
    int col = n_base + wn + j * 16 + (lane & 15);      // col = g*1024 + h
    #pragma unroll
    for (int i = 0; i < 2; ++i)
      #pragma unroll
      for (int r = 0; r < 8; ++r) {
        int b = wm + i * 16 + r + ((lane < 16) ? 0 : 8);
        // (g*B_+b)*H_+h == col_gate_block layout: g*BH + b*1024 + h
        int g = col >> 10, h = col & (H_ - 1);
        dst[((size_t)g * B_ + b) * H_ + h] = acc[i][j][r];
      }
  }
}

__device__ __forceinline__ float sigm(float x) { return 1.f / (1.f + __expf(-x)); }

// ---- per-step pointwise: reduce split-K partials + gx, gate math ----
__global__ void lstm_pointwise(const float* __restrict__ pp, const f16* __restrict__ gx_t,
                               float* __restrict__ c, float* __restrict__ h32,
                               f16* __restrict__ h16) {
  int i = blockIdx.x * 256 + threadIdx.x;    // BH
  float g4[4];
  #pragma unroll
  for (int g = 0; g < 4; ++g) {
    float s = (float)gx_t[(size_t)g * BH + i];
    #pragma unroll
    for (int kz = 0; kz < KSPLIT; ++kz)      // fixed order: deterministic
      s += pp[((size_t)kz * 4 + g) * BH + i];
    g4[g] = s;
  }
  float it = sigm(g4[0]);
  float ft = sigm(g4[1]);
  float gt = tanhf(g4[2]);
  float ot = sigm(g4[3]);
  float cv = ft * c[i] + it * gt;
  c[i] = cv;
  float hv = ot * tanhf(cv);
  h32[i] = hv;
  h16[i] = (f16)hv;
}

// ---- final FC: out[b][o] = h_T @ fcW^T + fcb (tiny, plain VALU) ----
__global__ void fc_kernel(const float* __restrict__ h32, const float* __restrict__ fcW,
                          const float* __restrict__ fcb, float* __restrict__ out) {
  int idx = blockIdx.x * 256 + threadIdx.x;
  if (idx >= B_ * O_) return;
  int b = idx / O_, o = idx % O_;
  const float* hr = h32 + (size_t)b * H_;
  const float* wr = fcW + (size_t)o * H_;
  float s = fcb[o];
  for (int k = 0; k < H_; ++k) s += hr[k] * wr[k];
  out[idx] = s;
}

extern "C" void kernel_launch(void* const* d_in, const int* in_sizes, int n_in,
                              void* d_out, int out_size, void* d_ws, size_t ws_size,
                              hipStream_t stream) {
  const int*   x   = (const int*)d_in[0];
  const float* emb = (const float*)d_in[1];
  const float* Wii = (const float*)d_in[2];
  const float* bii = (const float*)d_in[3];
  const float* Whi = (const float*)d_in[4];
  const float* Wif = (const float*)d_in[5];
  const float* bif = (const float*)d_in[6];
  const float* Whf = (const float*)d_in[7];
  const float* Wig = (const float*)d_in[8];
  const float* big = (const float*)d_in[9];
  const float* Whg = (const float*)d_in[10];
  const float* Wio = (const float*)d_in[11];
  const float* bio = (const float*)d_in[12];
  const float* Who = (const float*)d_in[13];
  const float* fcW = (const float*)d_in[14];
  const float* fcb = (const float*)d_in[15];
  float* out = (float*)d_out;

  // workspace carve-up (all chunk sizes multiples of 256B)
  char* w = (char*)d_ws;
  f16*   Wx16 = (f16*)w;   w += (size_t)4 * H_ * E_ * 2;        //   4 MB
  f16*   Wh16 = (f16*)w;   w += (size_t)4 * H_ * H_ * 2;        //   8 MB
  float* bx   = (float*)w; w += (size_t)N4H * 4;                //  16 KB
  f16*   e16  = (f16*)w;   w += (size_t)B_ * S_ * E_ * 2;       //  32 MB
  f16*   gx   = (f16*)w;   w += (size_t)S_ * 4 * B_ * H_ * 2;   // 256 MB
  float* pp   = (float*)w; w += (size_t)KSPLIT * 4 * BH * 4;    //   8 MB split-K partials
  float* cbuf = (float*)w; w += (size_t)BH * 4;
  float* h32  = (float*)w; w += (size_t)BH * 4;
  f16*   h16  = (f16*)w;   w += (size_t)BH * 2;

  // phase 0: pack / embed / init
  {
    int total = 4 * H_ * E_ + 4 * H_ * H_ + N4H;
    pack_kernel<<<(total + 255) / 256, 256, 0, stream>>>(
        Wii, Wif, Wig, Wio, Whi, Whf, Whg, Who, bii, bif, big, bio, Wx16, Wh16, bx);
  }
  embed_kernel<<<(B_ * S_ * E_) / 256, 256, 0, stream>>>(x, emb, e16);
  init_state_kernel<<<BH / 256, 256, 0, stream>>>(cbuf, h32, h16);

  // phase 1: all input-gate projections in one big WMMA GEMM
  gemm_x_kernel<<<dim3(N4H / 128, (B_ * S_) / 128), 256, 0, stream>>>(e16, Wx16, bx, gx);

  // phase 2: sequential LSTM scan (stream order serializes steps)
  for (int t = 0; t < S_; ++t) {
    const f16* gx_t = gx + (size_t)t * 4 * BH;
    gemm_h_kernel<<<dim3(N4H / 128, KSPLIT), 256, 0, stream>>>(h16, Wh16, pp);
    lstm_pointwise<<<BH / 256, 256, 0, stream>>>(pp, gx_t, cbuf, h32, h16);
  }

  // phase 3: final FC
  fc_kernel<<<(B_ * O_ + 255) / 256, 256, 0, stream>>>(h32, fcW, fcb, out);
}
